// MHA_21406117003857
// MI455X (gfx1250) — matrix operand order repaired and checked by reading, not compiled, not run on hardware
//
#include <hip/hip_runtime.h>

typedef unsigned short u16;
typedef __attribute__((ext_vector_type(16))) __bf16 v16bf;
typedef __attribute__((ext_vector_type(8)))  float  v8f;

constexpr int kN = 2048, kB = 4, kD = 512, kH = 8, kDH = 64, kInner = 512;
constexpr int kM = kN * kB;          // 8192 total rows
constexpr float kEps = 1e-5f;

__device__ __forceinline__ u16 f2bf(float f) {
  unsigned u = __float_as_uint(f);
  unsigned r = u + 0x7FFFu + ((u >> 16) & 1u);   // round-to-nearest-even
  return (u16)(r >> 16);
}

union FragCvt { uint4 u[2]; v16bf v; };

// A-fragment (16x32 bf16, MxK): lane half 0 holds K {k0+0..7, k0+16..23},
// half 1 holds K {k0+8..15, k0+24..31}; row = lane%16. Contiguous 16B loads.
__device__ __forceinline__ v16bf load_fragA(const u16* base, int stride, int row0, int k0) {
  int lane = threadIdx.x & 31, half = lane >> 4, m = lane & 15;
  const u16* p = base + (size_t)(row0 + m) * stride + k0 + half * 8;
  FragCvt f;
  f.u[0] = *(const uint4*)(p);
  f.u[1] = *(const uint4*)(p + 16);
  return f.v;
}
// B-fragment (32x16 bf16, KxN) from transposed (col-major-friendly) buffer:
// lane half h holds K {k0+8h..k0+8h+7, k0+16+8h..k0+16+8h+7}, column = lane%16.
__device__ __forceinline__ v16bf load_fragB(const u16* baseT, int stride, int col0, int k0) {
  int lane = threadIdx.x & 31, half = lane >> 4, n = lane & 15;
  const u16* p = baseT + (size_t)(col0 + n) * stride + k0 + half * 8;
  FragCvt f;
  f.u[0] = *(const uint4*)(p);
  f.u[1] = *(const uint4*)(p + 16);
  return f.v;
}

// ---- CDNA5 async global->LDS copy (ASYNCcnt path, ISA ch.10.7/15.18) ----
__device__ __forceinline__ void async_b128(const u16* lptr, const u16* gptr) {
  unsigned loff = (unsigned)(size_t)lptr;               // AS3 offset = low 32 bits
  unsigned long long ga = (unsigned long long)(size_t)gptr;
  asm volatile("global_load_async_to_lds_b128 %0, %1, off"
               :: "v"(loff), "v"(ga) : "memory");
}
__device__ __forceinline__ void wait_async0() {
  asm volatile("s_wait_asynccnt 0x0" ::: "memory");
}

__device__ __forceinline__ float half_sum(float v) {
  v += __shfl_xor(v, 1, 32); v += __shfl_xor(v, 2, 32);
  v += __shfl_xor(v, 4, 32); v += __shfl_xor(v, 8, 32);
  return v;
}
__device__ __forceinline__ float half_max(float v) {
  v = fmaxf(v, __shfl_xor(v, 1, 32)); v = fmaxf(v, __shfl_xor(v, 2, 32));
  v = fmaxf(v, __shfl_xor(v, 4, 32)); v = fmaxf(v, __shfl_xor(v, 8, 32));
  return v;
}

// ---------------- conversion / layout kernels ----------------
__global__ void cvt_bf16_kernel(const float* __restrict__ in, u16* __restrict__ out, int n) {
  int i = blockIdx.x * 256 + threadIdx.x;
  if (i < n) out[i] = f2bf(in[i]);
}
__global__ void transpose_cvt_kernel(const float* __restrict__ in, u16* __restrict__ out, int R, int C) {
  int i = blockIdx.x * 256 + threadIdx.x;
  if (i < R * C) { int r = i / C, c = i % C; out[(size_t)c * R + r] = f2bf(in[i]); }
}
// kvp (BH,N,DH) -> kvpT (BH,DH,N)
__global__ void kvpT_kernel(const u16* __restrict__ in, u16* __restrict__ out) {
  int i = blockIdx.x * 256 + threadIdx.x;
  int total = kB * kH * kN * kDH;
  if (i < total) {
    int dh = i % kDH; int n = (i / kDH) % kN; int bh = i / (kDH * kN);
    out[((size_t)bh * kDH + dh) * kN + n] = in[i];
  }
}

// ---------------- projection GEMM (+ optional per-head LayerNorm) ----------------
__global__ void __launch_bounds__(256)
proj_kernel(const u16* __restrict__ A, const u16* __restrict__ BT,
            u16* __restrict__ out, const float* __restrict__ g,
            const float* __restrict__ bta, int do_ln) {
  __shared__ u16 Abuf[2][128 * 32];   // 8KB per buffer
  __shared__ u16 Bbuf[2][64 * 32];    // 4KB per buffer
  int tid = threadIdx.x;
  int wave = tid >> 5, lane = tid & 31;
  int half = lane >> 4, nl = lane & 15;
  int rowsB = blockIdx.x * 128;
  int rows0 = rowsB + wave * 16;
  int head = blockIdx.y;
  int colB = head * 64;

  auto stage = [&](int k0, int buf) {
#pragma unroll
    for (int c = tid; c < 512; c += 256) {               // A: 128x32
      int row = c >> 2, part = c & 3;
      async_b128(&Abuf[buf][row * 32 + part * 8],
                 A + (size_t)(rowsB + row) * kD + k0 + part * 8);
    }
    {                                                    // B: 64x32
      int col = tid >> 2, part = tid & 3;
      async_b128(&Bbuf[buf][col * 32 + part * 8],
                 BT + (size_t)(colB + col) * kD + k0 + part * 8);
    }
  };

  v8f acc[4] = {};
  stage(0, 0);
  wait_async0();
  __syncthreads();
  for (int kc = 0; kc < 16; ++kc) {
    int buf = kc & 1;
    if (kc + 1 < 16) stage((kc + 1) * 32, buf ^ 1);      // prefetch next tile
    v16bf a = load_fragA(&Abuf[buf][0], 32, wave * 16, 0);
#pragma unroll
    for (int t = 0; t < 4; ++t) {
      v16bf bfr = load_fragB(&Bbuf[buf][0], 32, t * 16, 0);
      acc[t] = __builtin_amdgcn_wmma_f32_16x16x32_bf16(false, a, false, bfr,
                                                       (short)0, acc[t], false, false);
    }
    if (kc + 1 < 16) { wait_async0(); __syncthreads(); }
  }

  float gv[4] = {1.f,1.f,1.f,1.f}, bv[4] = {0.f,0.f,0.f,0.f};
  if (do_ln) {
#pragma unroll
    for (int t = 0; t < 4; ++t) { gv[t] = g[t * 16 + nl]; bv[t] = bta[t * 16 + nl]; }
  }
#pragma unroll
  for (int vg = 0; vg < 8; ++vg) {
    int r = rows0 + vg + 8 * half;
    int nseq = r >> 2, b = r & 3;          // r = n*B + b, B=4
    float mu = 0.f, inv = 1.f;
    if (do_ln) {
      float s1 = 0.f, s2 = 0.f;
#pragma unroll
      for (int t = 0; t < 4; ++t) { float v = acc[t][vg]; s1 += v; s2 += v * v; }
      s1 = half_sum(s1); s2 = half_sum(s2);
      mu = s1 * (1.f / 64.f);
      float var = s2 * (1.f / 64.f) - mu * mu;
      inv = rsqrtf(var + kEps);
    }
    size_t obase = ((size_t)(b * kH + head) * kN + nseq) * kDH;
#pragma unroll
    for (int t = 0; t < 4; ++t) {
      float v = acc[t][vg];
      if (do_ln) v = (v - mu) * inv * gv[t] + bv[t];
      out[obase + t * 16 + nl] = f2bf(v);
    }
  }
}

// ---------------- flash attention ----------------
// grid (B*H, N/256); 512 threads = 16 waves (4 per SIMD for latency hiding).
// K and V^T chunks of 64 keys staged in LDS via async copies, double-buffered.
// Each wave: 16 queries x DH=64, online softmax over 64-key chunks.
__global__ void __launch_bounds__(512)
fmha_kernel(const u16* __restrict__ qp, const u16* __restrict__ kvp,
            const u16* __restrict__ kvpT, u16* __restrict__ attn_out) {
  __shared__ u16 Kbuf[2][64 * 64];     // 8KB each: 64 keys x 64 dh
  __shared__ u16 Vtbuf[2][64 * 64];    // 8KB each: 64 dh x 64 keys
  __shared__ u16 plds[16 * 16 * 64];   // per-wave 16x64 P tile (32KB)
  int tid = threadIdx.x;
  int wave = tid >> 5, lane = tid & 31;
  int half = lane >> 4, nl = lane & 15;
  int bh = blockIdx.x;
  int b = bh / kH, h = bh % kH;
  int q0 = blockIdx.y * 256 + wave * 16;
  const u16* qBase = qp  + (size_t)bh * kN * kDH;
  const u16* kBase = kvp + (size_t)bh * kN * kDH;
  const u16* vT    = kvpT + (size_t)bh * kDH * kN;

  // 512 threads -> each issues 1 async b128 for K and 1 for V^T (8KB each)
  auto stage = [&](int j0, int buf) {
    int key = tid >> 3, part = tid & 7;                  // 64 x 64 keys-major
    async_b128(&Kbuf[buf][key * 64 + part * 8],
               kBase + (size_t)(j0 + key) * kDH + part * 8);
    int dh = key;                                        // 64 dh x 64 keys
    async_b128(&Vtbuf[buf][dh * 64 + part * 8],
               vT + (size_t)dh * kN + j0 + part * 8);
  };

  v16bf aq0 = load_fragA(qBase, kDH, q0, 0);
  v16bf aq1 = load_fragA(qBase, kDH, q0, 32);
  v8f oacc[4] = {};
  float mrun[8], lrun[8];
#pragma unroll
  for (int i = 0; i < 8; ++i) { mrun[i] = -1e30f; lrun[i] = 0.f; }
  u16* myP = plds + wave * 1024;

  constexpr int kChunks = kN / 64;
  stage(0, 0);
  wait_async0();
  __syncthreads();
  for (int jc = 0; jc < kChunks; ++jc) {
    int buf = jc & 1;
    if (jc + 1 < kChunks) stage((jc + 1) * 64, buf ^ 1); // prefetch next K/V
    v8f s[4] = {};
#pragma unroll
    for (int t = 0; t < 4; ++t) {                        // sim: Q(16x64) x K^T(64x64)
      v16bf bk0 = load_fragB(&Kbuf[buf][0], kDH, t * 16, 0);
      v16bf bk1 = load_fragB(&Kbuf[buf][0], kDH, t * 16, 32);
      s[t] = __builtin_amdgcn_wmma_f32_16x16x32_bf16(false, aq0, false, bk0,
                                                     (short)0, s[t], false, false);
      s[t] = __builtin_amdgcn_wmma_f32_16x16x32_bf16(false, aq1, false, bk1,
                                                     (short)0, s[t], false, false);
    }
#pragma unroll
    for (int vg = 0; vg < 8; ++vg) {                     // online softmax (64 keys)
      float sv[4];
      float mx = -1e30f;
#pragma unroll
      for (int t = 0; t < 4; ++t) { sv[t] = s[t][vg] * 0.125f; mx = fmaxf(mx, sv[t]); }
      mx = half_max(mx);
      float nm = fmaxf(mrun[vg], mx);
      float alpha = __expf(mrun[vg] - nm);
      int row = vg + 8 * half;
      float psum = 0.f;
#pragma unroll
      for (int t = 0; t < 4; ++t) {
        float pv = __expf(sv[t] - nm);
        psum += pv;
        myP[row * 64 + t * 16 + nl] = f2bf(pv);          // wave-private; DS in-order
      }
      psum = half_sum(psum);
      mrun[vg] = nm;
      lrun[vg] = lrun[vg] * alpha + psum;
#pragma unroll
      for (int t = 0; t < 4; ++t) oacc[t][vg] = oacc[t][vg] * alpha;
    }
    v16bf pa0 = load_fragA(myP, 64, 0, 0);
    v16bf pa1 = load_fragA(myP, 64, 0, 32);
#pragma unroll
    for (int t = 0; t < 4; ++t) {                        // O += P(16x64) x V(64x64)
      v16bf bv0 = load_fragB(&Vtbuf[buf][0], 64, t * 16, 0);
      v16bf bv1 = load_fragB(&Vtbuf[buf][0], 64, t * 16, 32);
      oacc[t] = __builtin_amdgcn_wmma_f32_16x16x32_bf16(false, pa0, false, bv0,
                                                        (short)0, oacc[t], false, false);
      oacc[t] = __builtin_amdgcn_wmma_f32_16x16x32_bf16(false, pa1, false, bv1,
                                                        (short)0, oacc[t], false, false);
    }
    if (jc + 1 < kChunks) { wait_async0(); __syncthreads(); }
  }
#pragma unroll
  for (int vg = 0; vg < 8; ++vg) {
    int nrow = q0 + vg + 8 * half;
    float linv = 1.f / lrun[vg];
    size_t obase = ((size_t)b * kN + nrow) * kInner + h * kDH;
#pragma unroll
    for (int t = 0; t < 4; ++t)
      attn_out[obase + t * 16 + nl] = f2bf(oacc[t][vg] * linv);
  }
}

// ---------------- output projection (+bias) -> fp32 scratch ----------------
__global__ void __launch_bounds__(256)
proj2_kernel(const u16* __restrict__ A, const u16* __restrict__ WoT,
             const float* __restrict__ bo, float* __restrict__ out) {
  __shared__ u16 Abuf[2][128 * 32];
  __shared__ u16 Bbuf[2][64 * 32];
  int tid = threadIdx.x;
  int wave = tid >> 5, lane = tid & 31;
  int half = lane >> 4, nl = lane & 15;
  int rowsB = blockIdx.x * 128;
  int rows0 = rowsB + wave * 16;
  int colB = blockIdx.y * 64;

  auto stage = [&](int k0, int buf) {
#pragma unroll
    for (int c = tid; c < 512; c += 256) {
      int row = c >> 2, part = c & 3;
      async_b128(&Abuf[buf][row * 32 + part * 8],
                 A + (size_t)(rowsB + row) * kInner + k0 + part * 8);
    }
    {
      int col = tid >> 2, part = tid & 3;
      async_b128(&Bbuf[buf][col * 32 + part * 8],
                 WoT + (size_t)(colB + col) * kInner + k0 + part * 8);
    }
  };

  v8f acc[4] = {};
  stage(0, 0);
  wait_async0();
  __syncthreads();
  for (int kc = 0; kc < 16; ++kc) {
    int buf = kc & 1;
    if (kc + 1 < 16) stage((kc + 1) * 32, buf ^ 1);
    v16bf a = load_fragA(&Abuf[buf][0], 32, wave * 16, 0);
#pragma unroll
    for (int t = 0; t < 4; ++t) {
      v16bf bfr = load_fragB(&Bbuf[buf][0], 32, t * 16, 0);
      acc[t] = __builtin_amdgcn_wmma_f32_16x16x32_bf16(false, a, false, bfr,
                                                       (short)0, acc[t], false, false);
    }
    if (kc + 1 < 16) { wait_async0(); __syncthreads(); }
  }
#pragma unroll
  for (int vg = 0; vg < 8; ++vg) {
    int r = rows0 + vg + 8 * half;
#pragma unroll
    for (int t = 0; t < 4; ++t) {
      int c = colB + t * 16 + nl;
      out[(size_t)r * kD + c] = acc[t][vg] + bo[c];
    }
  }
}

// ---------------- final LN * ls_scale, transpose to (N,B,D) ----------------
__global__ void __launch_bounds__(256)
lnout_kernel(const float* __restrict__ pin, const float* __restrict__ g,
             const float* __restrict__ bta, const float* __restrict__ ls,
             float* __restrict__ out) {
  int r = blockIdx.x;                 // r = b*N + n
  int b = r / kN, n = r % kN;
  const float* row = pin + (size_t)r * kD;
  float s1 = 0.f, s2 = 0.f;
  for (int i = threadIdx.x; i < kD; i += 256) { float v = row[i]; s1 += v; s2 += v * v; }
#pragma unroll
  for (int m = 1; m < 32; m <<= 1) { s1 += __shfl_xor(s1, m, 32); s2 += __shfl_xor(s2, m, 32); }
  __shared__ float sh1[8], sh2[8];
  int wave = threadIdx.x >> 5;
  if ((threadIdx.x & 31) == 0) { sh1[wave] = s1; sh2[wave] = s2; }
  __syncthreads();
  if (threadIdx.x == 0) {
    float a = 0.f, c = 0.f;
    for (int i = 0; i < 8; ++i) { a += sh1[i]; c += sh2[i]; }
    sh1[0] = a; sh2[0] = c;
  }
  __syncthreads();
  float mu = sh1[0] * (1.f / kD);
  float var = sh2[0] * (1.f / kD) - mu * mu;
  float inv = rsqrtf(var + kEps);
  float* orow = out + ((size_t)n * kB + b) * kD;
  for (int i = threadIdx.x; i < kD; i += 256)
    orow[i] = ((row[i] - mu) * inv * g[i] + bta[i]) * ls[i];
}

extern "C" void kernel_launch(void* const* d_in, const int* in_sizes, int n_in,
                              void* d_out, int out_size, void* d_ws, size_t ws_size,
                              hipStream_t stream) {
  (void)in_sizes; (void)n_in; (void)out_size; (void)ws_size;
  const float* x      = (const float*)d_in[0];
  const float* kv     = (const float*)d_in[1];
  const float* Wq     = (const float*)d_in[2];
  const float* Wkv    = (const float*)d_in[3];
  const float* ln_q_g = (const float*)d_in[4];
  const float* ln_q_b = (const float*)d_in[5];
  const float* Wo     = (const float*)d_in[6];
  const float* bo     = (const float*)d_in[7];
  const float* ln_o_g = (const float*)d_in[8];
  const float* ln_o_b = (const float*)d_in[9];
  const float* ls     = (const float*)d_in[10];
  float* out = (float*)d_out;

  char* ws = (char*)d_ws;
  size_t off = 0;
  auto alloc = [&](size_t bytes) { char* p = ws + off; off += (bytes + 255) & ~size_t(255); return p; };
  u16* xbf   = (u16*)alloc((size_t)kM * kD * 2);
  u16* kvbf  = (u16*)alloc((size_t)kM * kD * 2);
  u16* WqT   = (u16*)alloc((size_t)kD * kInner * 2);
  u16* WkvT  = (u16*)alloc((size_t)kD * kInner * 2);
  u16* WoT   = (u16*)alloc((size_t)kD * kInner * 2);
  u16* qp    = (u16*)alloc((size_t)kB * kH * kN * kDH * 2);
  u16* kvp   = (u16*)alloc((size_t)kB * kH * kN * kDH * 2);
  u16* kvpT  = (u16*)alloc((size_t)kB * kH * kDH * kN * 2);
  u16* aout  = (u16*)alloc((size_t)kM * kInner * 2);
  float* pws = (float*)alloc((size_t)kM * kD * 4);

  dim3 blk(256);
  int ne = kM * kD;
  cvt_bf16_kernel<<<dim3((ne + 255) / 256), blk, 0, stream>>>(x, xbf, ne);
  cvt_bf16_kernel<<<dim3((ne + 255) / 256), blk, 0, stream>>>(kv, kvbf, ne);
  int wn = kD * kInner;
  transpose_cvt_kernel<<<dim3((wn + 255) / 256), blk, 0, stream>>>(Wq, WqT, kD, kInner);
  transpose_cvt_kernel<<<dim3((wn + 255) / 256), blk, 0, stream>>>(Wkv, WkvT, kD, kInner);
  transpose_cvt_kernel<<<dim3((wn + 255) / 256), blk, 0, stream>>>(Wo, WoT, kInner, kD);

  proj_kernel<<<dim3(kM / 128, kH), blk, 0, stream>>>(xbf, WqT, qp, ln_q_g, ln_q_b, 1);
  proj_kernel<<<dim3(kM / 128, kH), blk, 0, stream>>>(kvbf, WkvT, kvp, ln_q_g, ln_q_b, 0);

  int tt = kB * kH * kN * kDH;
  kvpT_kernel<<<dim3((tt + 255) / 256), blk, 0, stream>>>(kvp, kvpT);

  fmha_kernel<<<dim3(kB * kH, kN / 256), dim3(512), 0, stream>>>(qp, kvp, kvpT, aout);

  proj2_kernel<<<dim3(kM / 128, kD / 64), blk, 0, stream>>>(aout, WoT, bo, pws);

  lnout_kernel<<<dim3(kM), blk, 0, stream>>>(pws, ln_o_g, ln_o_b, ls, out);
}
